// MultiHeadAttention_62955630625212
// MI455X (gfx1250) — matrix operand order repaired and checked by reading, not compiled, not run on hardware
//
#include <hip/hip_runtime.h>
#include <hip/hip_bf16.h>

typedef __bf16 bf16;
typedef __attribute__((ext_vector_type(16))) bf16 v16bf;
typedef __attribute__((ext_vector_type(8)))  bf16 v8bf;
typedef __attribute__((ext_vector_type(8)))  float v8f;

#define D_MODEL 1024
#define SEQ     2048
#define NB      2
#define NH      16
#define DK      64
#define M_TOTAL (NB * SEQ)   // 4096

__device__ __forceinline__ v16bf cat16(v8bf lo, v8bf hi) {
  v16bf r;
#pragma unroll
  for (int i = 0; i < 8; ++i) { r[i] = lo[i]; r[i + 8] = hi[i]; }
  return r;
}

__device__ __forceinline__ v8f vzero8() {
  v8f z;
#pragma unroll
  for (int i = 0; i < 8; ++i) z[i] = 0.0f;
  return z;
}

__device__ __forceinline__ v8f wmma_bf16(v16bf a, v16bf b, v8f c) {
  return __builtin_amdgcn_wmma_f32_16x16x32_bf16(
      /*neg_a=*/false, a, /*neg_b=*/false, b,
      /*c_mod=*/(short)0, c, /*reuse_a=*/false, /*reuse_b=*/false);
}

// ---------------------------------------------------------------------------
// Kernel 1: fp32 -> bf16 conversion (grid-stride)
// ---------------------------------------------------------------------------
__global__ __launch_bounds__(256) void cvt_f32_bf16(
    const float* __restrict__ in, bf16* __restrict__ out, int n) {
  for (int i = blockIdx.x * blockDim.x + threadIdx.x; i < n;
       i += gridDim.x * blockDim.x)
    out[i] = (bf16)in[i];
}

// ---------------------------------------------------------------------------
// Kernel 2: Y[4096,1024] = scale * (X(bf16) @ W(bf16)^T + bias)
// One wave computes a 32(M) x 64(N) tile: 2 A-frags, 4 B-frags shared across
// both M rows, 8 C-frags -> 8 WMMAs per 6KB of operands (~21 FLOP/B).
// mode 0: store bf16 as [B,H,S,DK]   (Q with scale=1/8, K)
// mode 1: store bf16 as [B,H,DK,S]   (V transposed)
// mode 2: store f32  as [4096,1024]  (final output)
// ---------------------------------------------------------------------------
__global__ __launch_bounds__(256) void gemm_xwT(
    const bf16* __restrict__ X, const bf16* __restrict__ W,
    const float* __restrict__ bias, float scale,
    bf16* __restrict__ outb, float* __restrict__ outf, int mode) {
  const int lane = threadIdx.x & 31;
  const int llo = lane & 15, lhi = lane >> 4;
  const int wid = blockIdx.x * (blockDim.x >> 5) + (threadIdx.x >> 5);
  const int ntiles = D_MODEL / 64;          // 16
  const int mtile = wid / ntiles;           // 0..127
  const int ntile = wid % ntiles;           // 0..15
  const int mbase = mtile * 32, nbase = ntile * 64;

  v8f acc[2][4];
#pragma unroll
  for (int i = 0; i < 2; ++i)
#pragma unroll
    for (int j = 0; j < 4; ++j) acc[i][j] = vzero8();

  const bf16* arow0 = X + (size_t)(mbase + llo) * D_MODEL;
  const bf16* arow1 = X + (size_t)(mbase + 16 + llo) * D_MODEL;
  for (int k0 = 0; k0 < D_MODEL; k0 += 32) {
    // A-fragments: lanes 0-15 hold K {k0..k0+15},
    // lanes 16-31 hold K {k0+16..31}.
    v16bf a0 = cat16(*(const v8bf*)(arow0 + k0 + lhi * 16),
                     *(const v8bf*)(arow0 + k0 + lhi * 16 + 8));
    v16bf a1 = cat16(*(const v8bf*)(arow1 + k0 + lhi * 16),
                     *(const v8bf*)(arow1 + k0 + lhi * 16 + 8));
#pragma unroll
    for (int j = 0; j < 4; ++j) {
      // B-fragment: col n = nbase+j*16+llo is row n of W (contiguous K).
      const bf16* wp =
          W + (size_t)(nbase + j * 16 + llo) * D_MODEL + k0 + lhi * 16;
      v16bf b = *(const v16bf*)wp;
      acc[0][j] = wmma_bf16(a0, b, acc[0][j]);
      acc[1][j] = wmma_bf16(a1, b, acc[1][j]);
    }
  }

#pragma unroll
  for (int j = 0; j < 4; ++j) {
    const int c = nbase + j * 16 + llo;
    const float bj = bias[c];
#pragma unroll
    for (int i = 0; i < 2; ++i) {
#pragma unroll
      for (int r = 0; r < 8; ++r) {
        const int gr = mbase + i * 16 + r + 8 * lhi;   // global row 0..4095
        const float val = (acc[i][j][r] + bj) * scale;
        if (mode == 2) {
          outf[(size_t)gr * D_MODEL + c] = val;
        } else {
          const int b = gr >> 11;             // batch
          const int s = gr & (SEQ - 1);       // seq pos
          const int h = c >> 6;               // head
          const int d = c & (DK - 1);         // dim in head
          if (mode == 0)
            outb[(((size_t)b * NH + h) * SEQ + s) * DK + d] = (bf16)val;
          else
            outb[(((size_t)b * NH + h) * DK + d) * SEQ + s] = (bf16)val;
        }
      }
    }
  }
}

// ---------------------------------------------------------------------------
// Kernel 3: flash attention. Block = 128 threads (4 waves); each wave owns 16
// query rows of one (b,h). Streams keys in 32-wide chunks with online softmax.
// Q was pre-scaled by 1/sqrt(DK) in its projection epilogue.
// Qh/Kh: [B,H,S,DK] bf16, Vt: [B,H,DK,S] bf16, Xo: [B,S,H*DK] bf16.
// ---------------------------------------------------------------------------
__global__ __launch_bounds__(128) void attn_kernel(
    const bf16* __restrict__ Qh, const bf16* __restrict__ Kh,
    const bf16* __restrict__ Vt, const int* __restrict__ mask,
    bf16* __restrict__ Xo) {
  __shared__ __align__(16) bf16 lds_p[4][16][40];  // +8 pad vs bank conflicts

  const int lane = threadIdx.x & 31;
  const int w = threadIdx.x >> 5;
  const int llo = lane & 15, lhi = lane >> 4;
  const int blk = blockIdx.x;       // bh * 32 + qt
  const int qt = blk & 31;          // 64-row query block index
  const int bh = blk >> 5;          // b*NH + h
  const int qbase = qt * 64 + w * 16;
  const size_t hbase = (size_t)bh * SEQ * DK;   // same size for Kh and Vt

  // Preload Q A-fragments for K-dim (= d) halves 0..31 and 32..63.
  v16bf qf[2];
#pragma unroll
  for (int jd = 0; jd < 2; ++jd) {
    const bf16* p = Qh + hbase + (size_t)(qbase + llo) * DK + jd * 32 + lhi * 16;
    qf[jd] = cat16(*(const v8bf*)p, *(const v8bf*)(p + 8));
  }

  float m_run[8], l_run[8];
  v8f acc[4];
#pragma unroll
  for (int r = 0; r < 8; ++r) { m_run[r] = -3.0e38f; l_run[r] = 0.0f; }
#pragma unroll
  for (int j = 0; j < 4; ++j) acc[j] = vzero8();

  for (int kc = 0; kc < SEQ; kc += 32) {
    // ---- scores: 16q x 32k, two 16x16 C-fragments, K-dim = 64 -------------
    v8f sc[2];
    sc[0] = vzero8(); sc[1] = vzero8();
#pragma unroll
    for (int t = 0; t < 2; ++t) {
#pragma unroll
      for (int jd = 0; jd < 2; ++jd) {
        const bf16* kp = Kh + hbase + (size_t)(kc + t * 16 + llo) * DK +
                         jd * 32 + lhi * 16;
        v16bf kf = *(const v16bf*)kp;
        sc[t] = wmma_bf16(qf[jd], kf, sc[t]);
      }
    }
    // ---- mask + online softmax (scores already scaled via Q) --------------
    float p0[8], p1[8];
#pragma unroll
    for (int r = 0; r < 8; ++r) {
      const int qs = qbase + r + 8 * lhi;
      float s0 = sc[0][r];
      float s1 = sc[1][r];
      if (mask[(size_t)qs * SEQ + kc + llo] == 0)      s0 = -1.0e9f;
      if (mask[(size_t)qs * SEQ + kc + 16 + llo] == 0) s1 = -1.0e9f;
      float mx = fmaxf(s0, s1);
#pragma unroll
      for (int off = 1; off < 16; off <<= 1)
        mx = fmaxf(mx, __shfl_xor(mx, off, 32));      // 16-lane row reduce
      const float nm = fmaxf(m_run[r], mx);
      const float corr = __expf(m_run[r] - nm);
      const float e0 = __expf(s0 - nm);
      const float e1 = __expf(s1 - nm);
      float rs = e0 + e1;
#pragma unroll
      for (int off = 1; off < 16; off <<= 1)
        rs += __shfl_xor(rs, off, 32);
      l_run[r] = l_run[r] * corr + rs;
      m_run[r] = nm;
      p0[r] = e0; p1[r] = e1;
#pragma unroll
      for (int j = 0; j < 4; ++j) acc[j][r] *= corr;
    }
    // ---- reshape P (C-frag layout) -> A-frag layout via LDS ---------------
#pragma unroll
    for (int r = 0; r < 8; ++r) {
      const int row = r + 8 * lhi;
      lds_p[w][row][llo] = (bf16)p0[r];
      lds_p[w][row][16 + llo] = (bf16)p1[r];
    }
    asm volatile("s_wait_dscnt 0x0" ::: "memory");
    const bf16* pp = &lds_p[w][llo][lhi * 16];
    v16bf pf = cat16(*(const v8bf*)pp, *(const v8bf*)(pp + 8));
    // ---- acc += P @ V (B-frag cols = d, K = keys; Vt contiguous in s) -----
#pragma unroll
    for (int j = 0; j < 4; ++j) {
      const bf16* vp =
          Vt + hbase + (size_t)(j * 16 + llo) * SEQ + kc + lhi * 16;
      v16bf vf = *(const v16bf*)vp;
      acc[j] = wmma_bf16(pf, vf, acc[j]);
    }
  }

  // ---- normalize + store [B,S,H*DK] bf16 ----------------------------------
  const int b = bh >> 4;
  const int h = bh & 15;
  const size_t obase = (size_t)b * SEQ * D_MODEL;
#pragma unroll
  for (int j = 0; j < 4; ++j) {
#pragma unroll
    for (int r = 0; r < 8; ++r) {
      const int qs = qbase + r + 8 * lhi;
      const float val = acc[j][r] / l_run[r];
      Xo[obase + (size_t)qs * D_MODEL + h * 64 + j * 16 + llo] = (bf16)val;
    }
  }
}

// ---------------------------------------------------------------------------
extern "C" void kernel_launch(void* const* d_in, const int* in_sizes, int n_in,
                              void* d_out, int out_size, void* d_ws,
                              size_t ws_size, hipStream_t stream) {
  const float* q    = (const float*)d_in[0];
  const float* k    = (const float*)d_in[1];
  const float* v    = (const float*)d_in[2];
  const int*   mask = (const int*)d_in[3];
  const float* Wq = (const float*)d_in[4];
  const float* bq = (const float*)d_in[5];
  const float* Wk = (const float*)d_in[6];
  const float* bk = (const float*)d_in[7];
  const float* Wv = (const float*)d_in[8];
  const float* bv = (const float*)d_in[9];
  const float* Wo = (const float*)d_in[10];
  const float* bo = (const float*)d_in[11];
  float* out = (float*)d_out;

  char* ws = (char*)d_ws;
  const size_t MB = 1024 * 1024;
  bf16* qb = (bf16*)(ws + 0 * MB);    // [4096,1024] bf16, 8MB each
  bf16* kb = (bf16*)(ws + 8 * MB);
  bf16* vb = (bf16*)(ws + 16 * MB);
  bf16* wq = (bf16*)(ws + 24 * MB);   // [1024,1024] bf16, 2MB each
  bf16* wk = (bf16*)(ws + 26 * MB);
  bf16* wv = (bf16*)(ws + 28 * MB);
  bf16* wo = (bf16*)(ws + 30 * MB);
  bf16* Qh = (bf16*)(ws + 32 * MB);   // [B,H,S,DK] bf16 (pre-scaled), 8MB
  bf16* Kh = (bf16*)(ws + 40 * MB);   // [B,H,S,DK]
  bf16* Vt = (bf16*)(ws + 48 * MB);   // [B,H,DK,S]
  bf16* Xo = (bf16*)(ws + 56 * MB);   // [B,S,D] bf16 attention output

  const int NX = M_TOTAL * D_MODEL;   // 4,194,304
  const int NW = D_MODEL * D_MODEL;   // 1,048,576

  cvt_f32_bf16<<<2048, 256, 0, stream>>>(q, qb, NX);
  cvt_f32_bf16<<<2048, 256, 0, stream>>>(k, kb, NX);
  cvt_f32_bf16<<<2048, 256, 0, stream>>>(v, vb, NX);
  cvt_f32_bf16<<<1024, 256, 0, stream>>>(Wq, wq, NW);
  cvt_f32_bf16<<<1024, 256, 0, stream>>>(Wk, wk, NW);
  cvt_f32_bf16<<<1024, 256, 0, stream>>>(Wv, wv, NW);
  cvt_f32_bf16<<<1024, 256, 0, stream>>>(Wo, wo, NW);

  // (4096/32 M-tiles) x (1024/64 N-tiles) = 2048 waves; 8 waves/block.
  const int gemm_blocks = (M_TOTAL / 32) * (D_MODEL / 64) / 8;  // 256
  const float qscale = 0.125f;        // 1/sqrt(DK), folded into Q projection
  gemm_xwT<<<gemm_blocks, 256, 0, stream>>>(qb, wq, bq, qscale, Qh, nullptr, 0);
  gemm_xwT<<<gemm_blocks, 256, 0, stream>>>(kb, wk, bk, 1.0f, Kh, nullptr, 0);
  gemm_xwT<<<gemm_blocks, 256, 0, stream>>>(vb, wv, bv, 1.0f, Vt, nullptr, 1);

  attn_kernel<<<NB * NH * (SEQ / 64), 128, 0, stream>>>(Qh, Kh, Vt, mask, Xo);

  gemm_xwT<<<gemm_blocks, 256, 0, stream>>>(Xo, wo, bo, 1.0f, nullptr, out, 2);
}
